// StructuredTransformer_73031623901636
// MI455X (gfx1250) — matrix operand order repaired and checked by reading, not compiled, not run on hardware
//
#include <hip/hip_runtime.h>
#include <math.h>

// ---------------- problem constants (match reference) ----------------
constexpr int NND   = 8192;   // N nodes
constexpr int KNB   = 16;     // K neighbors
constexpr int SZ    = 128;    // SIZE
constexpr int RELP  = 32;     // rel features padded 30 -> 32
constexpr int NHEAD = 8;
constexpr int ATTD  = 16;

typedef __attribute__((ext_vector_type(16))) _Float16 v16h;
typedef __attribute__((ext_vector_type(8)))  float    v8f;

union HFrag { v16h v; _Float16 h[16]; };

enum { M_DIRECT = 0, M_GATHER = 1, M_GATHER2 = 2 };

// ---------------------------------------------------------------------
// Weight pre-pack: B[Kd x Nc] (row-major, ldb, starting at row_off) ->
// WMMA B-fragment order, zero padded to Kp=KT*32, Ncp.
// Element e of lane's v16h corresponds to k = kt*32 + (lane>=16)*16 + e,
// n = nt*16 + (lane&15).  Linear layout: ((nt*KT + kt)*32 + lane)*16 + e.
// ---------------------------------------------------------------------
__global__ __launch_bounds__(256) void pack_b(
    const float* __restrict__ B, _Float16* __restrict__ Bp,
    int Kd, int Nc, int ldb, int row_off, int KT, int total)
{
  const int t = blockIdx.x * blockDim.x + threadIdx.x;
  if (t >= total) return;
  const int e    = t & 15;
  const int lane = (t >> 4) & 31;
  const int tile = t >> 9;
  const int kt   = tile % KT;
  const int nt   = tile / KT;
  const int k = kt * 32 + (lane >> 4) * 16 + e;
  const int n = nt * 16 + (lane & 15);
  float v = 0.f;
  if (k < Kd && n < Nc) v = B[(size_t)(row_off + k) * ldb + n];
  Bp[t] = (_Float16)v;
}

// ---------------------------------------------------------------------
// Tiled WMMA GEMM: C[M x Nc] = epilogue( rows(A)[M x 32*KT] @ Bp )
// Block = 256 threads = 8 waves; strip = 16 rows x up to 128 cols.
// Stage: the 16 x Kp f32 A-strip is loaded ONCE per block (2x b128 per
// thread), relu/cvt'd to f16 into padded LDS; each wave then builds its
// A fragment with 2x 16B ds loads per k-step.  B is prepacked f16
// fragments -> one 32B load per k-step.  No guards in the K loop.
// ---------------------------------------------------------------------
template <int KT, int MODE, bool RELU_A, bool RELU_OUT, bool ACCUM,
          bool HAS_BIAS, bool HAS_RESID>
__global__ __launch_bounds__(256) void gemm_wmma(
    const float* __restrict__ A0, const float* __restrict__ A1,
    const int*   __restrict__ idx,
    const _Float16* __restrict__ Bp, const float* __restrict__ bias,
    const float* __restrict__ resid, float* __restrict__ C,
    int lda, int ldc, int Nc, int Ncp)
{
  constexpr int KP     = KT * 32;
  constexpr int LDSROW = KP + 8;              // halves; keeps 16B align, skews banks
  __shared__ _Float16 As[16 * LDSROW];

  const int tid  = threadIdx.x;
  const int row0 = blockIdx.x * 16;

  // ---- cooperative stage: chunk = 8 floats -> 8 halves (16B) --------
  {
    constexpr int CPR = KP / 8;               // chunks per row
    constexpr int NCH = 16 * CPR;
    if (tid < NCH) {
      const int rr  = tid / CPR;
      const int cc  = (tid % CPR) * 8;
      const int row = row0 + rr;
      const float* arow;
      if (MODE == M_GATHER) {
        arow = A0 + (size_t)idx[row] * lda;
      } else if (MODE == M_GATHER2) {
        const int j = idx[row];
        arow = ((j < (row >> 4)) ? A0 : A1) + (size_t)j * lda;
      } else {
        arow = A0 + (size_t)row * lda;
      }
      const float4 f0 = *(const float4*)(arow + cc);
      const float4 f1 = *(const float4*)(arow + cc + 4);
      const float  rv[8] = {f0.x, f0.y, f0.z, f0.w, f1.x, f1.y, f1.z, f1.w};
      _Float16 hv[8];
#pragma unroll
      for (int i = 0; i < 8; ++i) {
        float v = rv[i];
        if (RELU_A) v = fmaxf(v, 0.f);
        hv[i] = (_Float16)v;
      }
      *(float4*)(&As[rr * LDSROW + cc]) = *(const float4*)hv;
    }
  }
  __syncthreads();

  // ---- per-wave 16x16 tile ------------------------------------------
  const int lane = tid & 31;
  const int wave = tid >> 5;                  // == n-tile index
  const int n0   = wave * 16;
  if (n0 < Ncp) {
    const int mn = lane & 15;
    const int hi = lane >> 4;
    const _Float16* bp = Bp + ((size_t)wave * KT * 32 + lane) * 16;
    const _Float16* as = &As[mn * LDSROW + hi * 16];

    v8f acc = {};
#pragma unroll
    for (int kt = 0; kt < KT; ++kt) {
      HFrag a;
      *(float4*)&a.h[0] = *(const float4*)(as + kt * 32);
      *(float4*)&a.h[8] = *(const float4*)(as + kt * 32 + 8);
      const v16h bv = *(const v16h*)(bp + (size_t)kt * 512);
      acc = __builtin_amdgcn_wmma_f32_16x16x32_f16(
          false, a.v, false, bv, (short)0, acc, false, false);
    }

    // epilogue: VGPR r holds (M = r + 8*hi, N = mn); branchless
    const int oc = n0 + mn;
    if (oc < Nc) {
      const float bval = HAS_BIAS ? bias[oc] : 0.f;
#pragma unroll
      for (int r = 0; r < 8; ++r) {
        const int    orow = row0 + r + hi * 8;
        const size_t off  = (size_t)orow * ldc + oc;
        float v = acc[r] + bval;
        if (ACCUM)     v += C[off];
        if (HAS_RESID) v += resid[off];
        if (RELU_OUT)  v = fmaxf(v, 0.f);
        C[off] = v;
      }
    }
  }
}

// ---------------------------------------------------------------------
// Relative geometric features, stride-32 padded rows:
// [rbf16 | dir3 | rot9 | sin | cos | 0 | 0]
// ---------------------------------------------------------------------
__global__ __launch_bounds__(256) void rel_features(
    const float* __restrict__ dist, const int* __restrict__ conn,
    float* __restrict__ rel)
{
  const int t = blockIdx.x * blockDim.x + threadIdx.x;
  if (t >= NND * KNB) return;
  const int n = t >> 4;
  const int j = conn[t];
  const float* xd = dist + (size_t)n * 13;
  const float* yd = dist + (size_t)j * 13;
  const float d0 = yd[0] - xd[0], d1 = yd[1] - xd[1], d2 = yd[2] - xd[2];
  const float dd   = sqrtf(d0 * d0 + d1 * d1 + d2 * d2 + 1e-8f);
  const float invd = 1.f / dd;
  float* rp = rel + (size_t)t * RELP;
  const float sig_inv = 16.f / 20.f;          // 1/sigma, sigma = 20/16
#pragma unroll
  for (int i = 0; i < 16; ++i) {              // centers = linspace(0,20,16)
    const float c = (20.f / 15.f) * (float)i;
    const float u = (dd - c) * sig_inv;
    rp[i] = __expf(-u * u);
  }
  rp[16] = d0 * invd; rp[17] = d1 * invd; rp[18] = d2 * invd;
#pragma unroll
  for (int i = 0; i < 3; ++i)                 // rot = xo^T * yo
#pragma unroll
    for (int l = 0; l < 3; ++l) {
      float s = 0.f;
#pragma unroll
      for (int jj = 0; jj < 3; ++jj)
        s += xd[3 + jj * 3 + i] * yd[3 + jj * 3 + l];
      rp[19 + i * 3 + l] = s;
    }
  const float dphi = xd[12] - yd[12];
  rp[28] = __sinf(dphi);
  rp[29] = __cosf(dphi);
  rp[30] = 0.f;
  rp[31] = 0.f;
}

// ---------------------------------------------------------------------
// Softmax attention over K=16 neighbors: thread = (node, head), float4 IO
// ---------------------------------------------------------------------
__global__ __launch_bounds__(256) void attn_softmax(
    const float* __restrict__ q, const float* __restrict__ kmat,
    const float* __restrict__ vmat, float* __restrict__ o)
{
  const int t = blockIdx.x * blockDim.x + threadIdx.x;
  if (t >= NND * NHEAD) return;
  const int n = t >> 3, h = t & 7;
  const float4* qp = (const float4*)(q + (size_t)n * SZ + h * ATTD);
  float4 qv[4];
#pragma unroll
  for (int a = 0; a < 4; ++a) qv[a] = qp[a];
  float s[KNB]; float mx = -1e30f;
#pragma unroll
  for (int k = 0; k < KNB; ++k) {
    const float4* kp = (const float4*)(kmat + ((size_t)n * KNB + k) * SZ + h * ATTD);
    float d = 0.f;
#pragma unroll
    for (int a = 0; a < 4; ++a) {
      const float4 kk = kp[a];
      d += qv[a].x * kk.x + qv[a].y * kk.y + qv[a].z * kk.z + qv[a].w * kk.w;
    }
    s[k] = d * 0.25f;                         // 1/sqrt(16)
    mx = fmaxf(mx, s[k]);
  }
  float sum = 0.f;
#pragma unroll
  for (int k = 0; k < KNB; ++k) { s[k] = __expf(s[k] - mx); sum += s[k]; }
  const float inv = 1.f / sum;
  float4 ov[4] = {{0,0,0,0},{0,0,0,0},{0,0,0,0},{0,0,0,0}};
#pragma unroll
  for (int k = 0; k < KNB; ++k) {
    const float w = s[k] * inv;
    const float4* vp = (const float4*)(vmat + ((size_t)n * KNB + k) * SZ + h * ATTD);
#pragma unroll
    for (int a = 0; a < 4; ++a) {
      const float4 vv = vp[a];
      ov[a].x += w * vv.x; ov[a].y += w * vv.y;
      ov[a].z += w * vv.z; ov[a].w += w * vv.w;
    }
  }
  float4* op = (float4*)(o + (size_t)n * SZ + h * ATTD);
#pragma unroll
  for (int a = 0; a < 4; ++a) op[a] = ov[a];
}

// ---------------------------------------------------------------------
// Decoder seq-onehot contribution (float4 per thread):
// C[r,:] += (conn[r] < node) ? Wseq[seq[conn[r]],:] : 0
// ---------------------------------------------------------------------
__global__ __launch_bounds__(256) void seq_add(
    float* __restrict__ Cm, const float* __restrict__ Wseq,
    const int* __restrict__ seqv, const int* __restrict__ conn)
{
  const int t = blockIdx.x * blockDim.x + threadIdx.x;
  if (t >= NND * KNB * (SZ / 4)) return;
  const int r    = t >> 5;                    // row in [0, N*K)
  const int c4   = (t & 31);                  // float4 column
  const int j    = conn[r];
  const int node = r >> 4;
  if (j >= node) return;
  float4*       cp = (float4*)(Cm + (size_t)r * SZ) + c4;
  const float4  wv = ((const float4*)(Wseq + (size_t)seqv[j] * SZ))[c4];
  float4 v = *cp;
  v.x += wv.x; v.y += wv.y; v.z += wv.z; v.w += wv.w;
  *cp = v;
}

// ---------------------------------------------------------------------
extern "C" void kernel_launch(void* const* d_in, const int* in_sizes, int n_in,
                              void* d_out, int out_size, void* d_ws, size_t ws_size,
                              hipStream_t stream)
{
  (void)in_sizes; (void)n_in; (void)out_size; (void)ws_size;
  const float* features  = (const float*)d_in[0];
  const int*   sequence  = (const int*)  d_in[1];
  const float* distances = (const float*)d_in[2];
  const int*   conn      = (const int*)  d_in[3];

  // params flattened pytree-style (dict keys alphabetical):
  // dec[0..2]{att{Wk,Wo,Wq,Wv,bo}, mlp[(W,b)x4]}, enc[0..2]{...}, post[W,b], pre[W,b]
  struct Lay { const float *Wk,*Wo,*Wq,*Wv,*bo,*mW[4],*mb[4]; };
  Lay dec[3], enc[3];
  int p = 4;
  auto grab = [&](Lay& L) {
    L.Wk = (const float*)d_in[p++]; L.Wo = (const float*)d_in[p++];
    L.Wq = (const float*)d_in[p++]; L.Wv = (const float*)d_in[p++];
    L.bo = (const float*)d_in[p++];
    for (int i = 0; i < 4; ++i) {
      L.mW[i] = (const float*)d_in[p++];
      L.mb[i] = (const float*)d_in[p++];
    }
  };
  for (int l = 0; l < 3; ++l) grab(dec[l]);
  for (int l = 0; l < 3; ++l) grab(enc[l]);
  const float* postW = (const float*)d_in[p++];
  const float* postb = (const float*)d_in[p++];
  const float* preW  = (const float*)d_in[p++];
  const float* preb  = (const float*)d_in[p++];

  // ---------------- workspace: f32 buffers then packed f16 weights ----
  float* w     = (float*)d_ws;
  float* rel   = w; w += (size_t)NND * KNB * RELP;
  float* x     = w; w += (size_t)NND * SZ;
  float* encb  = w; w += (size_t)NND * SZ;
  float* h1    = w; w += (size_t)NND * SZ;
  float* h2    = w; w += (size_t)NND * SZ;
  float* local = w; w += (size_t)NND * SZ;
  float* q     = w; w += (size_t)NND * SZ;
  float* o     = w; w += (size_t)NND * SZ;
  float* kmat  = w; w += (size_t)NND * KNB * SZ;
  float* vmat  = w; w += (size_t)NND * KNB * SZ;
  _Float16* hcur = (_Float16*)w;

  const dim3 blk(256);
  auto pack = [&](const float* B, int Kd, int Nc, int ldb, int row_off) -> _Float16* {
    const int KT    = (Kd + 31) / 32;
    const int Ncp   = (Nc + 15) & ~15;
    const int total = KT * 32 * Ncp;
    _Float16* dst = hcur; hcur += total;
    pack_b<<<(total + 255) / 256, blk, 0, stream>>>(B, dst, Kd, Nc, ldb, row_off, KT, total);
    return dst;
  };

  struct PLay { _Float16 *mW[4], *Wq, *Wo, *Wk_l, *Wk_r, *Wv_l, *Wv_r; };
  PLay pe[3], pd[3];
  for (int l = 0; l < 3; ++l) {
    for (int i = 0; i < 4; ++i) pe[l].mW[i] = pack(enc[l].mW[i], SZ, SZ, SZ, 0);
    pe[l].Wq   = pack(enc[l].Wq, SZ, SZ, SZ, 0);
    pe[l].Wo   = pack(enc[l].Wo, SZ, SZ, SZ, 0);
    pe[l].Wk_l = pack(enc[l].Wk, SZ, SZ, SZ, 0);
    pe[l].Wk_r = pack(enc[l].Wk, 30, SZ, SZ, 128);
    pe[l].Wv_l = pack(enc[l].Wv, SZ, SZ, SZ, 0);
    pe[l].Wv_r = pack(enc[l].Wv, 30, SZ, SZ, 128);
  }
  for (int l = 0; l < 3; ++l) {
    for (int i = 0; i < 4; ++i) pd[l].mW[i] = pack(dec[l].mW[i], SZ, SZ, SZ, 0);
    pd[l].Wq   = pack(dec[l].Wq, SZ, SZ, SZ, 0);
    pd[l].Wo   = pack(dec[l].Wo, SZ, SZ, SZ, 0);
    pd[l].Wk_l = pack(dec[l].Wk, SZ, SZ, SZ, 0);
    pd[l].Wk_r = pack(dec[l].Wk, 30, SZ, SZ, 128);
    pd[l].Wv_l = pack(dec[l].Wv, SZ, SZ, SZ, 0);
    pd[l].Wv_r = pack(dec[l].Wv, 30, SZ, SZ, 128);
  }
  _Float16* preWp  = pack(preW,  64, SZ, SZ, 0);
  _Float16* postWp = pack(postW, SZ, 20, 20, 0);

  const int gN  = NND / 16;                   // 512 row-strips
  const int gNK = (NND * KNB) / 16;           // 8192 row-strips

  rel_features<<<(NND * KNB + 255) / 256, blk, 0, stream>>>(distances, conn, rel);

  // x = features @ Wpre + b        (KT=2)
  gemm_wmma<2, M_DIRECT, false, false, false, true, false><<<gN, blk, 0, stream>>>(
      features, nullptr, nullptr, preWp, preb, nullptr, x, 64, SZ, SZ, SZ);

  // ------------------------------ encoder ---------------------------
  for (int l = 0; l < 3; ++l) {
    const Lay& L = enc[l]; const PLay& P = pe[l];
    gemm_wmma<4, M_DIRECT, true, true, false, true, false><<<gN, blk, 0, stream>>>(
        x,  nullptr, nullptr, P.mW[0], L.mb[0], nullptr, h1, SZ, SZ, SZ, SZ);
    gemm_wmma<4, M_DIRECT, false, true, false, true, false><<<gN, blk, 0, stream>>>(
        h1, nullptr, nullptr, P.mW[1], L.mb[1], nullptr, h2, SZ, SZ, SZ, SZ);
    gemm_wmma<4, M_DIRECT, false, true, false, true, false><<<gN, blk, 0, stream>>>(
        h2, nullptr, nullptr, P.mW[2], L.mb[2], nullptr, h1, SZ, SZ, SZ, SZ);
    gemm_wmma<4, M_DIRECT, false, true, false, true, false><<<gN, blk, 0, stream>>>(
        h1, nullptr, nullptr, P.mW[3], L.mb[3], nullptr, local, SZ, SZ, SZ, SZ);
    gemm_wmma<4, M_DIRECT, false, false, false, false, false><<<gN, blk, 0, stream>>>(
        local, nullptr, nullptr, P.Wq, nullptr, nullptr, q, SZ, SZ, SZ, SZ);
    // k = gather(local)[conn] @ Wk[:128] + rel @ Wk[128:158]
    gemm_wmma<4, M_GATHER, false, false, false, false, false><<<gNK, blk, 0, stream>>>(
        local, nullptr, conn, P.Wk_l, nullptr, nullptr, kmat, SZ, SZ, SZ, SZ);
    gemm_wmma<1, M_DIRECT, false, false, true, false, false><<<gNK, blk, 0, stream>>>(
        rel, nullptr, nullptr, P.Wk_r, nullptr, nullptr, kmat, RELP, SZ, SZ, SZ);
    gemm_wmma<4, M_GATHER, false, false, false, false, false><<<gNK, blk, 0, stream>>>(
        local, nullptr, conn, P.Wv_l, nullptr, nullptr, vmat, SZ, SZ, SZ, SZ);
    gemm_wmma<1, M_DIRECT, false, false, true, false, false><<<gNK, blk, 0, stream>>>(
        rel, nullptr, nullptr, P.Wv_r, nullptr, nullptr, vmat, RELP, SZ, SZ, SZ);
    attn_softmax<<<(NND * NHEAD + 255) / 256, blk, 0, stream>>>(q, kmat, vmat, o);
    gemm_wmma<4, M_DIRECT, false, false, false, true, true><<<gN, blk, 0, stream>>>(
        o, nullptr, nullptr, P.Wo, L.bo, x, x, SZ, SZ, SZ, SZ);  // x += o@Wo+bo
  }

  hipMemcpyAsync(encb, x, (size_t)NND * SZ * sizeof(float),
                 hipMemcpyDeviceToDevice, stream);

  // ------------------------------ decoder ---------------------------
  for (int l = 0; l < 3; ++l) {
    const Lay& L = dec[l]; const PLay& P = pd[l];
    gemm_wmma<4, M_DIRECT, true, true, false, true, false><<<gN, blk, 0, stream>>>(
        x,  nullptr, nullptr, P.mW[0], L.mb[0], nullptr, h1, SZ, SZ, SZ, SZ);
    gemm_wmma<4, M_DIRECT, false, true, false, true, false><<<gN, blk, 0, stream>>>(
        h1, nullptr, nullptr, P.mW[1], L.mb[1], nullptr, h2, SZ, SZ, SZ, SZ);
    gemm_wmma<4, M_DIRECT, false, true, false, true, false><<<gN, blk, 0, stream>>>(
        h2, nullptr, nullptr, P.mW[2], L.mb[2], nullptr, h1, SZ, SZ, SZ, SZ);
    gemm_wmma<4, M_DIRECT, false, true, false, true, false><<<gN, blk, 0, stream>>>(
        h1, nullptr, nullptr, P.mW[3], L.mb[3], nullptr, local, SZ, SZ, SZ, SZ);
    gemm_wmma<4, M_DIRECT, false, false, false, false, false><<<gN, blk, 0, stream>>>(
        local, nullptr, nullptr, P.Wq, nullptr, nullptr, q, SZ, SZ, SZ, SZ);
    // k = (pre ? local[conn] : enc[conn]) @ Wk[:128] + rel @ Wk[128:158] + onehot
    gemm_wmma<4, M_GATHER2, false, false, false, false, false><<<gNK, blk, 0, stream>>>(
        local, encb, conn, P.Wk_l, nullptr, nullptr, kmat, SZ, SZ, SZ, SZ);
    gemm_wmma<1, M_DIRECT, false, false, true, false, false><<<gNK, blk, 0, stream>>>(
        rel, nullptr, nullptr, P.Wk_r, nullptr, nullptr, kmat, RELP, SZ, SZ, SZ);
    seq_add<<<(NND * KNB * (SZ / 4)) / 256, blk, 0, stream>>>(
        kmat, L.Wk + 158 * SZ, sequence, conn);
    gemm_wmma<4, M_GATHER2, false, false, false, false, false><<<gNK, blk, 0, stream>>>(
        local, encb, conn, P.Wv_l, nullptr, nullptr, vmat, SZ, SZ, SZ, SZ);
    gemm_wmma<1, M_DIRECT, false, false, true, false, false><<<gNK, blk, 0, stream>>>(
        rel, nullptr, nullptr, P.Wv_r, nullptr, nullptr, vmat, RELP, SZ, SZ, SZ);
    seq_add<<<(NND * KNB * (SZ / 4)) / 256, blk, 0, stream>>>(
        vmat, L.Wv + 158 * SZ, sequence, conn);
    attn_softmax<<<(NND * NHEAD + 255) / 256, blk, 0, stream>>>(q, kmat, vmat, o);
    gemm_wmma<4, M_DIRECT, false, false, false, true, true><<<gN, blk, 0, stream>>>(
        o, nullptr, nullptr, P.Wo, L.bo, x, x, SZ, SZ, SZ, SZ);
  }

  // out = y @ Wpost + b   (Nc = 20 ragged, packed Ncp = 32)
  gemm_wmma<4, M_DIRECT, false, false, false, true, false><<<gN, blk, 0, stream>>>(
      x, nullptr, nullptr, postWp, postb, nullptr, (float*)d_out, SZ, 20, 20, 32);
}
